// LLaDA2MoeDecoderLayer_27161373179913
// MI455X (gfx1250) — matrix operand order repaired and checked by reading, not compiled, not run on hardware
//
#include <hip/hip_runtime.h>
#include <hip/hip_bf16.h>

// ---------------- problem constants ----------------
#define S_   1024
#define H_   2048
#define NH_  16
#define NKV_ 4
#define HD_  128
#define E_   16
#define I_   1024
#define EPS_ 1e-6f

typedef __attribute__((ext_vector_type(16))) __bf16 v16bf;
typedef __attribute__((ext_vector_type(8)))  __bf16 v8bf;
typedef __attribute__((ext_vector_type(8)))  float  v8f;

// ---------------- bf16 helpers (native converts -> v_cvt_pk_bf16_f32) ----------------
static __device__ __forceinline__ __bf16 f2bf(float f) { return (__bf16)f; }
static __device__ __forceinline__ unsigned short f2bfbits(float f) {
    return __builtin_bit_cast(unsigned short, (__bf16)f);
}
static __device__ __forceinline__ v16bf mk16(v8bf lo, v8bf hi) {
    v16bf r;
#pragma unroll
    for (int e = 0; e < 8; ++e) { r[e] = lo[e]; r[8 + e] = hi[e]; }
    return r;
}

#define WMMA_BF16(a, b, c) \
    __builtin_amdgcn_wmma_f32_16x16x32_bf16(false, (a), false, (b), (short)0, (c), false, false)

// LDS tile row length (32 K-elements + 8 pad -> 80B stride, 16B aligned)
#define TROW 40

// ---------------- staging primitives ----------------
// Convert+store 8 f32 -> 8 bf16 into LDS.
static __device__ __forceinline__ void cvt8_store(const float* __restrict__ src,
                                                  unsigned short* dst) {
    float t[8];
    *(float4*)(t + 0) = ((const float4*)src)[0];
    *(float4*)(t + 4) = ((const float4*)src)[1];
    v8bf v;
#pragma unroll
    for (int e = 0; e < 8; ++e) v[e] = f2bf(t[e]);
    *(v8bf*)dst = v;
}
// CDNA5 async 16-byte global->LDS copy (tracked by ASYNCcnt).
static __device__ __forceinline__ void async_cp16(unsigned int ldsoff,
                                                  const unsigned short* src) {
    asm volatile("global_load_async_to_lds_b128 %0, %1, off"
                 :: "v"(ldsoff), "v"((unsigned long long)(size_t)src)
                 : "memory");
}
static __device__ __forceinline__ void async_wait() {
    asm volatile("s_wait_asynccnt 0x0" ::: "memory");
}

// Fragment loads from LDS tiles (layouts per CDNA5 ISA 7.12.2).
static __device__ __forceinline__ v16bf fragA(const unsigned short (*As)[TROW], int row, int lane) {
    int base = (lane & 16) ? 8 : 0;
    v8bf lo = *(const v8bf*)&As[row][base];
    v8bf hi = *(const v8bf*)&As[row][base + 16];
    return mk16(lo, hi);
}
static __device__ __forceinline__ v16bf fragB(const unsigned short (*Bs)[TROW], int col, int lane) {
    int base = (lane & 16) ? 8 : 0;
    v8bf lo = *(const v8bf*)&Bs[col][base];
    v8bf hi = *(const v8bf*)&Bs[col][base + 16];
    return mk16(lo, hi);
}

// ---------------- kernel 0: zero expert counters ----------------
__global__ void zero_counts_kernel(int* cnt) {
    if (threadIdx.x < E_) cnt[threadIdx.x] = 0;
}

// ---------------- kernel 1: RMSNorm over H (f32 and/or bf16 outputs) ----------------
__global__ __launch_bounds__(256) void rmsnorm_kernel(const float* __restrict__ x,
                                                      const float* __restrict__ w,
                                                      float* __restrict__ yf,
                                                      unsigned short* __restrict__ ybf) {
    __shared__ float red[256];
    int row = blockIdx.x;
    int t = threadIdx.x;
    const float* xr = x + (size_t)row * H_;
    float s = 0.f;
#pragma unroll
    for (int i = 0; i < H_ / 256; ++i) {
        float v = xr[t + i * 256];
        s += v * v;
    }
    red[t] = s;
    __syncthreads();
    for (int off = 128; off > 0; off >>= 1) {
        if (t < off) red[t] += red[t + off];
        __syncthreads();
    }
    float r = rsqrtf(red[0] * (1.0f / H_) + EPS_);
#pragma unroll
    for (int i = 0; i < H_ / 256; ++i) {
        int c = t + i * 256;
        float v = w[c] * xr[c] * r;
        if (yf)  yf[(size_t)row * H_ + c] = v;
        if (ybf) ybf[(size_t)row * H_ + c] = f2bfbits(v);
    }
}

// ---------------- kernel 2: LDS-tiled WMMA GEMM, bf16 A (async), f32 B ----------------
// C[M,N] = A[M,K] * B[N,K]^T (+Res). Block = 256 thr = 8 waves -> 128x64 tile.
__global__ __launch_bounds__(256) void gemm_bf16_kernel(const unsigned short* __restrict__ A,
                                                        const float* __restrict__ B,
                                                        const float* __restrict__ Res,
                                                        float* __restrict__ C,
                                                        int M, int N, int Kd) {
    __shared__ unsigned short As[128][TROW];
    __shared__ unsigned short Bs[64][TROW];
    int tid = threadIdx.x;
    int lane = tid & 31;
    int wv = tid >> 5;
    int tilesN = N >> 6;
    int mb = blockIdx.x / tilesN;
    int n0 = (blockIdx.x % tilesN) << 6;

    // per-thread staging pointers (hoisted out of K loop)
    int arow = tid >> 1, apart = tid & 1;
    int am = mb * 128 + arow; if (am >= M) am = M - 1;
    const unsigned short* a_src = A + (size_t)am * Kd + apart * 16;
    unsigned int a_lds = (unsigned int)(size_t)&As[arow][apart * 16];
    int bcol = tid >> 2, bq = tid & 3;
    const float* b_src = B + (size_t)(n0 + bcol) * Kd + bq * 8;
    unsigned short* b_dst = &Bs[bcol][bq * 8];

    v8f acc[4] = {v8f{}, v8f{}, v8f{}, v8f{}};
    for (int k0 = 0; k0 < Kd; k0 += 32) {
        async_cp16(a_lds, a_src + k0);
        async_cp16(a_lds + 16, a_src + k0 + 8);
        if (k0 + 32 < Kd) __builtin_prefetch(b_src + k0 + 32, 0, 0);
        cvt8_store(b_src + k0, b_dst);
        async_wait();
        __syncthreads();
        v16bf a = fragA(As, wv * 16 + (lane & 15), lane);
        v16bf b0 = fragB(Bs, 0 * 16 + (lane & 15), lane);
        v16bf b1 = fragB(Bs, 1 * 16 + (lane & 15), lane);
        v16bf b2 = fragB(Bs, 2 * 16 + (lane & 15), lane);
        v16bf b3 = fragB(Bs, 3 * 16 + (lane & 15), lane);
        acc[0] = WMMA_BF16(a, b0, acc[0]);
        acc[1] = WMMA_BF16(a, b1, acc[1]);
        acc[2] = WMMA_BF16(a, b2, acc[2]);
        acc[3] = WMMA_BF16(a, b3, acc[3]);
        __syncthreads();
    }
    int hi = (lane & 16) ? 8 : 0;
#pragma unroll
    for (int j = 0; j < 4; ++j) {
        int col = n0 + j * 16 + (lane & 15);
#pragma unroll
        for (int r = 0; r < 8; ++r) {
            int m = mb * 128 + wv * 16 + r + hi;
            if (m < M) {
                float v = acc[j][r];
                if (Res) v += Res[(size_t)m * N + col];
                C[(size_t)m * N + col] = v;
            }
        }
    }
}

// ---------------- kernel 3: per-head RMSNorm + RoPE; bf16 Q/K rows, bf16 V transposed ----------------
__global__ __launch_bounds__(128) void qkv_post_kernel(const float* __restrict__ qkv,
                                                       const float* __restrict__ cosb,
                                                       const float* __restrict__ sinb,
                                                       const float* __restrict__ qw,
                                                       const float* __restrict__ kw,
                                                       unsigned short* __restrict__ qb,
                                                       unsigned short* __restrict__ kb,
                                                       unsigned short* __restrict__ vbT) {
    __shared__ float red[128];
    __shared__ float qs[128];
    int s = blockIdx.x;
    int t = threadIdx.x;
    const float* row = qkv + (size_t)s * (NH_ + 2 * NKV_) * HD_;
    float cs = cosb[s * HD_ + t];
    float sn = sinb[s * HD_ + t];

    for (int h = 0; h < NH_ + NKV_; ++h) {
        bool isQ = (h < NH_);
        float v = row[h * HD_ + t];
        red[t] = v * v;
        __syncthreads();
        for (int off = 64; off > 0; off >>= 1) {
            if (t < off) red[t] += red[t + off];
            __syncthreads();
        }
        float r = rsqrtf(red[0] * (1.0f / HD_) + EPS_);
        float n = (isQ ? qw[t] : kw[t]) * v * r;
        qs[t] = n;
        __syncthreads();
        float rot = (t < HD_ / 2) ? -qs[t + HD_ / 2] : qs[t - HD_ / 2];
        float out = n * cs + rot * sn;
        if (isQ)
            qb[((size_t)h * S_ + s) * HD_ + t] = f2bfbits(out);
        else
            kb[((size_t)(h - NH_) * S_ + s) * HD_ + t] = f2bfbits(out);
        __syncthreads();
    }
    // V: transpose to [kvh][d][s] so attention B-fragments are contiguous
#pragma unroll
    for (int h = 0; h < NKV_; ++h)
        vbT[((size_t)h * HD_ + t) * S_ + s] = f2bfbits(row[(NH_ + NKV_ + h) * HD_ + t]);
}

// ---------------- kernel 4: attention (one wave per head x 16-row Q tile) ----------------
__global__ __launch_bounds__(32) void attention_kernel(const unsigned short* __restrict__ qb,
                                                       const unsigned short* __restrict__ kb,
                                                       const unsigned short* __restrict__ vbT,
                                                       unsigned short* __restrict__ attn_out) {
    __shared__ __bf16 sc[16][S_ + 8];                // probability buffer
    int head = blockIdx.x >> 6;
    int qt = blockIdx.x & 63;
    int lane = threadIdx.x;
    int kvh = head >> 2;                             // GQA 4:1
    const float scale = 0.08838834764831845f;        // 1/sqrt(128)
    int hi = (lane & 16) ? 8 : 0;

    const unsigned short* qrow = qb + ((size_t)head * S_ + qt * 16 + (lane & 15)) * HD_;

    // ---- pass 1: scores = Q K^T ----
    for (int j = 0; j < S_ / 16; ++j) {
        const unsigned short* kcol = kb + ((size_t)kvh * S_ + j * 16 + (lane & 15)) * HD_;
        v8f acc = v8f{};
#pragma unroll
        for (int c = 0; c < HD_ / 32; ++c) {
            int ab = c * 32 + ((lane & 16) ? 8 : 0);
            v16bf a = mk16(*(const v8bf*)(qrow + ab), *(const v8bf*)(qrow + ab + 16));
            int bb = c * 32 + ((lane & 16) ? 8 : 0);
            v16bf b = mk16(*(const v8bf*)(kcol + bb), *(const v8bf*)(kcol + bb + 16));
            acc = WMMA_BF16(a, b, acc);
        }
#pragma unroll
        for (int r = 0; r < 8; ++r)
            sc[r + hi][j * 16 + (lane & 15)] = f2bf(acc[r] * scale);
    }
    __syncthreads();

    // ---- softmax (one row per lane 0..15) ----
    if (lane < 16) {
        float m = -3.4e38f;
        for (int c = 0; c < S_; ++c) m = fmaxf(m, (float)sc[lane][c]);
        float sum = 0.f;
        for (int c = 0; c < S_; ++c) {
            float e = __expf((float)sc[lane][c] - m);
            sc[lane][c] = f2bf(e);
            sum += e;
        }
        float inv = 1.0f / sum;
        for (int c = 0; c < S_; ++c) sc[lane][c] = f2bf((float)sc[lane][c] * inv);
    }
    __syncthreads();

    // ---- pass 2: out = P V (V pre-transposed, contiguous fragments) ----
    v8f oacc[8];
#pragma unroll
    for (int j = 0; j < 8; ++j) oacc[j] = v8f{};
    const __bf16* prow = &sc[lane & 15][0];
    const unsigned short* vcol[8];
#pragma unroll
    for (int jc = 0; jc < 8; ++jc)
        vcol[jc] = vbT + ((size_t)kvh * HD_ + jc * 16 + (lane & 15)) * S_;

    for (int kv0 = 0; kv0 < S_; kv0 += 32) {
        int ab = kv0 + ((lane & 16) ? 8 : 0);
        v16bf a = mk16(*(const v8bf*)(prow + ab), *(const v8bf*)(prow + ab + 16));
        int bb = kv0 + ((lane & 16) ? 8 : 0);
#pragma unroll
        for (int jc = 0; jc < 8; ++jc) {
            v16bf b = mk16(*(const v8bf*)(vcol[jc] + bb), *(const v8bf*)(vcol[jc] + bb + 16));
            oacc[jc] = WMMA_BF16(a, b, oacc[jc]);
        }
    }
#pragma unroll
    for (int jc = 0; jc < 8; ++jc) {
        int col = head * HD_ + jc * 16 + (lane & 15);
#pragma unroll
        for (int r = 0; r < 8; ++r) {
            int m = qt * 16 + r + hi;
            attn_out[(size_t)m * H_ + col] = f2bfbits(oacc[jc][r]);
        }
    }
}

// ---------------- kernel 5: gating ----------------
__global__ __launch_bounds__(256) void gating_kernel(const float* __restrict__ xp,
                                                     const float* __restrict__ gw,
                                                     int* __restrict__ ecnt,
                                                     int* __restrict__ etok,
                                                     float* __restrict__ ewt) {
    __shared__ float red[256];
    __shared__ float logits[E_];
    int s = blockIdx.x;
    int t = threadIdx.x;
    int e = t >> 4;
    int p = t & 15;
    const float* xr = xp + (size_t)s * H_;
    const float* wr = gw + (size_t)e * H_;
    float acc = 0.f;
    for (int c = p; c < H_; c += 16) acc += xr[c] * wr[c];
    red[t] = acc;
    __syncthreads();
    if (t < E_) {
        float sum = 0.f;
#pragma unroll
        for (int q = 0; q < 16; ++q) sum += red[t * 16 + q];
        logits[t] = sum;
    }
    __syncthreads();
    if (t == 0) {
        float m = -3.4e38f;
#pragma unroll
        for (int i = 0; i < E_; ++i) m = fmaxf(m, logits[i]);
        float probs[E_];
        float z = 0.f;
#pragma unroll
        for (int i = 0; i < E_; ++i) { probs[i] = __expf(logits[i] - m); z += probs[i]; }
        float inv = 1.0f / z;
#pragma unroll
        for (int i = 0; i < E_; ++i) probs[i] *= inv;
        int i1 = 0;
#pragma unroll
        for (int i = 1; i < E_; ++i) if (probs[i] > probs[i1]) i1 = i;
        int i2 = (i1 == 0) ? 1 : 0;
#pragma unroll
        for (int i = 0; i < E_; ++i) if (i != i1 && probs[i] > probs[i2]) i2 = i;
        float v1 = probs[i1], v2 = probs[i2];
        float invs = 1.0f / (v1 + v2);
        int pos1 = atomicAdd(&ecnt[i1], 1);
        etok[i1 * S_ + pos1] = s;
        ewt[i1 * S_ + pos1] = v1 * invs;
        int pos2 = atomicAdd(&ecnt[i2], 1);
        etok[i2 * S_ + pos2] = s;
        ewt[i2 * S_ + pos2] = v2 * invs;
    }
}

// ---------------- kernel 6: MoE fc1 (gathered async bf16 A, dual f32 B) ----------------
__global__ __launch_bounds__(256) void moe_fc1_kernel(const unsigned short* __restrict__ xpb,
                                                      const float* __restrict__ w1,
                                                      const float* __restrict__ w2,
                                                      const int* __restrict__ etok,
                                                      const int* __restrict__ ecnt,
                                                      unsigned short* __restrict__ act) {
    __shared__ unsigned short As[128][TROW];
    __shared__ unsigned short Bs1[64][TROW];
    __shared__ unsigned short Bs2[64][TROW];
    int tid = threadIdx.x;
    int lane = tid & 31;
    int wv = tid >> 5;
    int e = blockIdx.x >> 7;
    int rem = blockIdx.x & 127;
    int rt = rem >> 4;                // 8 row tiles of 128
    int n0 = (rem & 15) << 6;         // 16 strips of 64 over I
    int cnt = ecnt[e];
    if (rt * 128 >= cnt) return;

    int arow = tid >> 1, apart = tid & 1;
    int slot = rt * 128 + arow; if (slot >= cnt) slot = cnt - 1;
    int tok = etok[e * S_ + slot];
    const unsigned short* a_src = xpb + (size_t)tok * H_ + apart * 16;
    unsigned int a_lds = (unsigned int)(size_t)&As[arow][apart * 16];
    int bcol = tid >> 2, bq = tid & 3;
    const float* b1_src = w1 + (size_t)e * I_ * H_ + (size_t)(n0 + bcol) * H_ + bq * 8;
    const float* b2_src = w2 + (size_t)e * I_ * H_ + (size_t)(n0 + bcol) * H_ + bq * 8;
    unsigned short* b1_dst = &Bs1[bcol][bq * 8];
    unsigned short* b2_dst = &Bs2[bcol][bq * 8];

    v8f a1[4] = {v8f{}, v8f{}, v8f{}, v8f{}};
    v8f a2[4] = {v8f{}, v8f{}, v8f{}, v8f{}};
    for (int k0 = 0; k0 < H_; k0 += 32) {
        async_cp16(a_lds, a_src + k0);
        async_cp16(a_lds + 16, a_src + k0 + 8);
        if (k0 + 32 < H_) __builtin_prefetch(b1_src + k0 + 32, 0, 0);
        cvt8_store(b1_src + k0, b1_dst);
        cvt8_store(b2_src + k0, b2_dst);
        async_wait();
        __syncthreads();
        v16bf a = fragA(As, wv * 16 + (lane & 15), lane);
#pragma unroll
        for (int j = 0; j < 4; ++j) {
            v16bf f1 = fragB(Bs1, j * 16 + (lane & 15), lane);
            a1[j] = WMMA_BF16(a, f1, a1[j]);
            v16bf f2 = fragB(Bs2, j * 16 + (lane & 15), lane);
            a2[j] = WMMA_BF16(a, f2, a2[j]);
        }
        __syncthreads();
    }
    int hi = (lane & 16) ? 8 : 0;
#pragma unroll
    for (int j = 0; j < 4; ++j) {
        int col = n0 + j * 16 + (lane & 15);
#pragma unroll
        for (int r = 0; r < 8; ++r) {
            int sl = rt * 128 + wv * 16 + r + hi;
            if (sl < cnt) {
                float h1 = a1[j][r];
                float h2 = a2[j][r];
                float silu = h1 * (1.0f / (1.0f + __expf(-h1)));
                act[((size_t)e * S_ + sl) * I_ + col] = f2bfbits(silu * h2);
            }
        }
    }
}

// ---------------- kernel 7: MoE fc2 (async bf16 A) + weighted scatter ----------------
__global__ __launch_bounds__(256) void moe_fc2_kernel(const unsigned short* __restrict__ act,
                                                      const float* __restrict__ w3,
                                                      const int* __restrict__ etok,
                                                      const float* __restrict__ ewt,
                                                      const int* __restrict__ ecnt,
                                                      float* __restrict__ out) {
    __shared__ unsigned short As[128][TROW];
    __shared__ unsigned short Bs[64][TROW];
    int tid = threadIdx.x;
    int lane = tid & 31;
    int wv = tid >> 5;
    int e = blockIdx.x >> 8;
    int rem = blockIdx.x & 255;
    int rt = rem >> 5;                // 8 row tiles of 128
    int n0 = (rem & 31) << 6;         // 32 strips of 64 over H
    int cnt = ecnt[e];
    if (rt * 128 >= cnt) return;

    int arow = tid >> 1, apart = tid & 1;
    int slot = rt * 128 + arow; if (slot >= cnt) slot = cnt - 1;
    const unsigned short* a_src = act + ((size_t)e * S_ + slot) * I_ + apart * 16;
    unsigned int a_lds = (unsigned int)(size_t)&As[arow][apart * 16];
    int bcol = tid >> 2, bq = tid & 3;
    const float* b_src = w3 + (size_t)e * H_ * I_ + (size_t)(n0 + bcol) * I_ + bq * 8;
    unsigned short* b_dst = &Bs[bcol][bq * 8];

    v8f acc[4] = {v8f{}, v8f{}, v8f{}, v8f{}};
    for (int k0 = 0; k0 < I_; k0 += 32) {
        async_cp16(a_lds, a_src + k0);
        async_cp16(a_lds + 16, a_src + k0 + 8);
        if (k0 + 32 < I_) __builtin_prefetch(b_src + k0 + 32, 0, 0);
        cvt8_store(b_src + k0, b_dst);
        async_wait();
        __syncthreads();
        v16bf a = fragA(As, wv * 16 + (lane & 15), lane);
#pragma unroll
        for (int j = 0; j < 4; ++j) {
            v16bf b = fragB(Bs, j * 16 + (lane & 15), lane);
            acc[j] = WMMA_BF16(a, b, acc[j]);
        }
        __syncthreads();
    }
    int hi = (lane & 16) ? 8 : 0;
    int tokr[8];
    float wtr[8];
#pragma unroll
    for (int r = 0; r < 8; ++r) {
        int sl = rt * 128 + wv * 16 + r + hi;
        if (sl < cnt) { tokr[r] = etok[e * S_ + sl]; wtr[r] = ewt[e * S_ + sl]; }
        else { tokr[r] = -1; wtr[r] = 0.f; }
    }
#pragma unroll
    for (int j = 0; j < 4; ++j) {
        int col = n0 + j * 16 + (lane & 15);
#pragma unroll
        for (int r = 0; r < 8; ++r) {
            if (tokr[r] >= 0)
                atomicAdd(&out[(size_t)tokr[r] * H_ + col], acc[j][r] * wtr[r]);
        }
    }
}

// ---------------- host launcher ----------------
extern "C" void kernel_launch(void* const* d_in, const int* in_sizes, int n_in,
                              void* d_out, int out_size, void* d_ws, size_t ws_size,
                              hipStream_t stream) {
    (void)in_sizes; (void)n_in; (void)out_size; (void)ws_size;
    const float* hidden  = (const float*)d_in[0];
    const float* cosb    = (const float*)d_in[1];
    const float* sinb    = (const float*)d_in[2];
    const float* qkv_w   = (const float*)d_in[3];
    const float* q_ln_w  = (const float*)d_in[4];
    const float* k_ln_w  = (const float*)d_in[5];
    const float* dense_w = (const float*)d_in[6];
    const float* in_ln_w = (const float*)d_in[7];
    const float* post_ln = (const float*)d_in[8];
    const float* gate_w  = (const float*)d_in[9];
    const float* fc1_1_w = (const float*)d_in[10];
    const float* fc1_2_w = (const float*)d_in[11];
    const float* fc2_w   = (const float*)d_in[12];
    float* out = (float*)d_out;

    const size_t M1 = 1u << 20;   // float units
    float* ws = (float*)d_ws;
    unsigned short* xnb  = (unsigned short*)(ws);            // S*H bf16   (1M f)
    float* qkv           = ws + 1 * M1;                      // S*3072 f32 (3M f)
    unsigned short* qbf  = (unsigned short*)(ws + 4 * M1);   // NH*S*HD bf16 (1M f)
    unsigned short* kbf  = (unsigned short*)(ws + 5 * M1);   // NKV*S*HD bf16 (256K f)
    unsigned short* vbT  = (unsigned short*)(ws + 5 * M1 + 262144);  // (256K f)
    unsigned short* attb = (unsigned short*)(ws + 6 * M1);   // S*H bf16 (1M f)
    float* x_post        = ws + 7 * M1;                      // S*H f32 (2M f)
    unsigned short* xpb  = (unsigned short*)(ws + 9 * M1);   // S*H bf16 (1M f)
    int*   ecnt          = (int*)(ws + 10 * M1);             // 16 (+pad)
    int*   etok          = ecnt + 32;                        // E*S
    float* ewt           = (float*)(etok + E_ * S_);         // E*S
    unsigned short* act  = (unsigned short*)(ws + 11 * M1);  // E*S*I bf16 (8M f)

    zero_counts_kernel<<<1, 32, 0, stream>>>(ecnt);
    rmsnorm_kernel<<<S_, 256, 0, stream>>>(hidden, in_ln_w, nullptr, xnb);
    // QKV: (1024/128)*(3072/64) = 384 blocks
    gemm_bf16_kernel<<<384, 256, 0, stream>>>(xnb, qkv_w, nullptr, qkv, S_, (NH_ + 2 * NKV_) * HD_, H_);
    qkv_post_kernel<<<S_, 128, 0, stream>>>(qkv, cosb, sinb, q_ln_w, k_ln_w, qbf, kbf, vbT);
    attention_kernel<<<NH_ * (S_ / 16), 32, 0, stream>>>(qbf, kbf, vbT, attb);
    // dense + residual: (1024/128)*(2048/64) = 256 blocks
    gemm_bf16_kernel<<<256, 256, 0, stream>>>(attb, dense_w, hidden, out, S_, H_, H_);
    rmsnorm_kernel<<<S_, 256, 0, stream>>>(out, post_ln, x_post, xpb);
    gating_kernel<<<S_, 256, 0, stream>>>(x_post, gate_w, ecnt, etok, ewt);
    // fc1: 16 experts * 8 row tiles * 16 strips = 2048 blocks
    moe_fc1_kernel<<<2048, 256, 0, stream>>>(xpb, fc1_1_w, fc1_2_w, etok, ecnt, act);
    // fc2: 16 * 8 * 32 = 4096 blocks
    moe_fc2_kernel<<<4096, 256, 0, stream>>>(act, fc2_w, etok, ewt, ecnt, out);
}
